// GINModel_59193239273690
// MI455X (gfx1250) — compile-verified, matrix operand order checked by reading, not run on hardware
//
#include <hip/hip_runtime.h>
#include <hip/hip_bf16.h>
#include <stdint.h>

#define N_NODES 50000
#define N_EDGES 800000
#define IN_CH   64
#define HIDDEN  128
#define OUT_CH  64

typedef __attribute__((ext_vector_type(16))) __bf16 v16bf;
typedef __attribute__((ext_vector_type(8)))  float  v8f;

// K position held by VGPR v (0..7, i.e. bf16 pair index) for a lane in the
// given half (lane>>4), per CDNA5 ISA 16-bit A/B fragment layout:
//   lanes 0-15 : V0..V3 -> K=0..7,  V4..V7 -> K=16..23
//   lanes 16-31: V0..V3 -> K=8..15, V4..V7 -> K=24..31
__device__ __forceinline__ int kmap(int v, int half) {
    return (v < 4 ? 2 * v : 8 + 2 * v) + 8 * half;
}

__global__ void k_zero(float* __restrict__ p, long n) {
    long i = (long)blockIdx.x * blockDim.x + threadIdx.x;
    if (i < n) p[i] = 0.0f;
}

// Pre-pack a row-major [K][Nf] f32 weight matrix into bf16 WMMA B-fragments:
// packed[((kc*NT + nt)*32 + lane)*16 + j], j = 2*v + lo, so each lane's
// 32-byte fragment is contiguous (two dwordx4 loads in the GEMM kernel).
template <int K, int Nf>
__global__ void k_pack_w(const float* __restrict__ W, __bf16* __restrict__ out) {
    const int NT = Nf / 16;
    int p = blockIdx.x * blockDim.x + threadIdx.x;
    if (p >= K * Nf) return;
    int j    = p & 15;
    int lane = (p >> 4) & 31;
    int rest = p >> 9;
    int nt   = rest % NT;
    int kc   = rest / NT;
    int v  = j >> 1, lo = j & 1;
    int half = lane >> 4;
    int k = kc * 32 + kmap(v, half) + lo;
    int n = nt * 16 + (lane & 15);
    out[p] = (__bf16)W[(size_t)k * Nf + n];
}

// Edge-parallel scatter-add: agg[dst] += feat[src]. float4 gather + 4 f32
// atomics; working set (feat + agg) is L2-resident on MI455X (192MB L2).
template <int C>
__global__ void k_scatter(const float* __restrict__ feat, const int* __restrict__ src,
                          const int* __restrict__ dst, float* __restrict__ agg) {
    const int G = C / 4;
    long gid = (long)blockIdx.x * blockDim.x + threadIdx.x;
    long e = gid / G;
    int  g = (int)(gid % G);
    if (e >= N_EDGES) return;
    int s = src[e], d = dst[e];
    const float4 val = *(const float4*)(feat + (size_t)s * C + 4 * g);
    float* a = agg + (size_t)d * C + 4 * g;
    atomicAdd(a + 0, val.x);
    atomicAdd(a + 1, val.y);
    atomicAdd(a + 2, val.z);
    atomicAdd(a + 3, val.w);
}

// Fused GIN MLP: h0 = (1+eps)*x + agg ; h1 = relu(h0 @ Wa + ba) ;
// out = h1 @ Wb + bb (optional relu). Block = 4 waves, each wave owns a
// 16-row M-tile; GEMMs via v_wmma_f32_16x16x32_bf16 with f32 accumulate.
template <int CIN, int CHID, int COUT, bool RELU_OUT>
__global__ __launch_bounds__(128) void k_gin_mlp(
    const float* __restrict__ x, const float* __restrict__ agg,
    const float* __restrict__ epsp,
    const __bf16* __restrict__ pWa, const float* __restrict__ ba,
    const __bf16* __restrict__ pWb, const float* __restrict__ bb,
    float* __restrict__ out) {
    const int NT1 = CHID / 16, NT2 = COUT / 16;
    const int KC1 = CIN / 32,  KC2 = CHID / 32;

    __shared__ __bf16 sA[4 * 16 * CIN];     // per-wave bf16 input tiles
    __shared__ __bf16 sMid[4 * 16 * CHID];  // per-wave bf16 hidden tiles

    const int wave  = threadIdx.x >> 5;
    const int lane  = threadIdx.x & 31;
    const int half  = lane >> 4;
    const int idx16 = lane & 15;
    const int nodeBase = blockIdx.x * 64 + wave * 16;
    const float scale = 1.0f + epsp[0];

    __bf16* sAw = sA + wave * (16 * CIN);
    __bf16* sMw = sMid + wave * (16 * CHID);

    // Stage 0: combine self + aggregated neighbors, convert to bf16.
    for (int i = lane; i < 16 * CIN; i += 32) {
        int r = i / CIN, c = i % CIN;
        int m = nodeBase + r;
        float v = 0.0f;
        if (m < N_NODES) {
            size_t off = (size_t)m * CIN + c;
            v = scale * x[off] + agg[off];
        }
        sAw[r * CIN + c] = (__bf16)v;
    }
    __syncthreads();

    union AF { v16bf v; uint32_t u[8]; };
    union BF { v16bf v; uint4 q[2]; };

    // GEMM1: [16 x CIN] @ [CIN x CHID]
    v8f acc[NT1];
    #pragma unroll
    for (int nt = 0; nt < NT1; ++nt) { v8f z = {}; acc[nt] = z; }

    for (int kc = 0; kc < KC1; ++kc) {
        AF a;
        #pragma unroll
        for (int v = 0; v < 8; ++v) {
            int k = kc * 32 + kmap(v, half);   // even -> aligned b32 LDS read
            a.u[v] = *(const uint32_t*)(sAw + idx16 * CIN + k);
        }
        #pragma unroll
        for (int nt = 0; nt < NT1; ++nt) {
            BF b;
            const __bf16* pb = pWa + (size_t)(kc * NT1 + nt) * 512 + lane * 16;
            b.q[0] = *(const uint4*)(pb);
            b.q[1] = *(const uint4*)(pb + 8);
            acc[nt] = __builtin_amdgcn_wmma_f32_16x16x32_bf16(
                false, a.v, false, b.v, (short)0, acc[nt], false, false);
        }
    }

    // Bias + ReLU, write hidden tile back to LDS as bf16 (C/D layout:
    // lane<16 holds rows 0..7, lane>=16 rows 8..15).
    #pragma unroll
    for (int nt = 0; nt < NT1; ++nt) {
        #pragma unroll
        for (int r = 0; r < 8; ++r) {
            int m = half * 8 + r;
            int n = nt * 16 + idx16;
            float val = acc[nt][r] + ba[n];
            val = fmaxf(val, 0.0f);
            sMw[m * CHID + n] = (__bf16)val;
        }
    }
    __syncthreads();

    // GEMM2: [16 x CHID] @ [CHID x COUT]
    v8f acc2[NT2];
    #pragma unroll
    for (int nt = 0; nt < NT2; ++nt) { v8f z = {}; acc2[nt] = z; }

    for (int kc = 0; kc < KC2; ++kc) {
        AF a;
        #pragma unroll
        for (int v = 0; v < 8; ++v) {
            int k = kc * 32 + kmap(v, half);
            a.u[v] = *(const uint32_t*)(sMw + idx16 * CHID + k);
        }
        #pragma unroll
        for (int nt = 0; nt < NT2; ++nt) {
            BF b;
            const __bf16* pb = pWb + (size_t)(kc * NT2 + nt) * 512 + lane * 16;
            b.q[0] = *(const uint4*)(pb);
            b.q[1] = *(const uint4*)(pb + 8);
            acc2[nt] = __builtin_amdgcn_wmma_f32_16x16x32_bf16(
                false, a.v, false, b.v, (short)0, acc2[nt], false, false);
        }
    }

    // Bias (+optional ReLU), store f32.
    #pragma unroll
    for (int nt = 0; nt < NT2; ++nt) {
        #pragma unroll
        for (int r = 0; r < 8; ++r) {
            int m = half * 8 + r;
            int n = nt * 16 + idx16;
            int mg = nodeBase + m;
            float val = acc2[nt][r] + bb[n];
            if (RELU_OUT) val = fmaxf(val, 0.0f);
            if (mg < N_NODES) out[(size_t)mg * COUT + n] = val;
        }
    }
}

extern "C" void kernel_launch(void* const* d_in, const int* in_sizes, int n_in,
                              void* d_out, int out_size, void* d_ws, size_t ws_size,
                              hipStream_t stream) {
    const float* x  = (const float*)d_in[0];
    const int*   ei = (const int*)d_in[1];
    const float* W1 = (const float*)d_in[2];
    const float* b1 = (const float*)d_in[3];
    const float* W2 = (const float*)d_in[4];
    const float* b2 = (const float*)d_in[5];
    const float* e1 = (const float*)d_in[6];
    const float* W3 = (const float*)d_in[7];
    const float* b3 = (const float*)d_in[8];
    const float* W4 = (const float*)d_in[9];
    const float* b4 = (const float*)d_in[10];
    const float* e2 = (const float*)d_in[11];
    const int* src = ei;
    const int* dst = ei + N_EDGES;
    float* outp = (float*)d_out;

    char* ws = (char*)d_ws;
    size_t off = 0;
    auto alloc = [&](size_t bytes) -> char* {
        char* p = ws + off;
        off += (bytes + 255) & ~(size_t)255;
        return p;
    };

    float*  agg1 = (float*)alloc((size_t)N_NODES * IN_CH * 4);
    float*  agg2 = (float*)alloc((size_t)N_NODES * HIDDEN * 4);
    float*  h1   = (float*)alloc((size_t)N_NODES * HIDDEN * 4);
    __bf16* pW1  = (__bf16*)alloc((size_t)IN_CH * HIDDEN * 2);
    __bf16* pW2  = (__bf16*)alloc((size_t)HIDDEN * HIDDEN * 2);
    __bf16* pW3  = (__bf16*)alloc((size_t)HIDDEN * HIDDEN * 2);
    __bf16* pW4  = (__bf16*)alloc((size_t)HIDDEN * OUT_CH * 2);

    // Zero aggregation buffers (ws is not re-poisoned between replays).
    {
        long n = (long)N_NODES * IN_CH;
        k_zero<<<(unsigned)((n + 255) / 256), 256, 0, stream>>>(agg1, n);
        n = (long)N_NODES * HIDDEN;
        k_zero<<<(unsigned)((n + 255) / 256), 256, 0, stream>>>(agg2, n);
    }

    // Pack weights into bf16 WMMA B-fragment layout.
    k_pack_w<IN_CH, HIDDEN><<<(IN_CH * HIDDEN + 255) / 256, 256, 0, stream>>>(W1, pW1);
    k_pack_w<HIDDEN, HIDDEN><<<(HIDDEN * HIDDEN + 255) / 256, 256, 0, stream>>>(W2, pW2);
    k_pack_w<HIDDEN, HIDDEN><<<(HIDDEN * HIDDEN + 255) / 256, 256, 0, stream>>>(W3, pW3);
    k_pack_w<HIDDEN, OUT_CH><<<(HIDDEN * OUT_CH + 255) / 256, 256, 0, stream>>>(W4, pW4);

    // Layer 1
    {
        long n = (long)N_EDGES * (IN_CH / 4);
        k_scatter<IN_CH><<<(unsigned)((n + 255) / 256), 256, 0, stream>>>(x, src, dst, agg1);
        k_gin_mlp<IN_CH, HIDDEN, HIDDEN, true>
            <<<(N_NODES + 63) / 64, 128, 0, stream>>>(x, agg1, e1, pW1, b1, pW2, b2, h1);
    }
    // Layer 2
    {
        long n = (long)N_EDGES * (HIDDEN / 4);
        k_scatter<HIDDEN><<<(unsigned)((n + 255) / 256), 256, 0, stream>>>(h1, src, dst, agg2);
        k_gin_mlp<HIDDEN, HIDDEN, OUT_CH, false>
            <<<(N_NODES + 63) / 64, 128, 0, stream>>>(h1, agg2, e2, pW3, b3, pW4, b4, outp);
    }
}
